// ConditionalBlock_54674933678402
// MI455X (gfx1250) — matrix operand order repaired and checked by reading, not compiled, not run on hardware
//
#include <hip/hip_runtime.h>
#include <math.h>

// ---------------------------------------------------------------------------
// DiT conditional block for MI455X (gfx1250, wave32, WMMA 16x16x32 bf16,
// async global->LDS staging on the ASYNCcnt path).
// Pipeline:
//   Wt  = bf16 transpose of each weight  (once per launch, L2-resident after)
//   mod = silu(cond) @ Wada + bada
//   xm  = modulate(LN(x))  -> bf16
//   qkv = xm @ Wqkv        -> bf16        (async-staged WMMA GEMM, mode 0)
//   att = flash-attention(qkv) -> bf16    (WMMA attention, async Q/K staging)
//   x1  = x + gate_a*(att @ Wproj + bproj) -> f32   (mode 1)
//   xm2 = modulate(LN(x1)) -> bf16
//   h   = gelu(xm2 @ W1 + b1) -> bf16               (mode 2)
//   out = x1 + gate_m*(h @ W2 + b2) -> f32          (mode 1)
// ---------------------------------------------------------------------------

#define DD   1024
#define TT   1024
#define BB   4
#define HH   16
#define HDIM 64
#define MLPH 4096
#define BT   (BB * TT)
#define MODN (6 * DD)
#define EPS  1e-5f

typedef __bf16        v16bf __attribute__((ext_vector_type(16)));
typedef float         v8f   __attribute__((ext_vector_type(8)));
typedef unsigned int  u32x4 __attribute__((ext_vector_type(4)));

union FragAB {           // 32 bytes: one WMMA 16-bit A or B operand (16 bf16)
    v16bf v;
    u32x4 q[2];
};

__device__ inline v8f v8f_zero() {
    v8f z = {0.f, 0.f, 0.f, 0.f, 0.f, 0.f, 0.f, 0.f};
    return z;
}

__device__ inline unsigned short f32_to_bf16(float f) {
    union { float f; unsigned u; } cv; cv.f = f;
    unsigned u = cv.u;
    u += 0x7FFFu + ((u >> 16) & 1u);   // round-to-nearest-even
    return (unsigned short)(u >> 16);
}

__device__ inline v8f wmma_bf16(const FragAB& a, const FragAB& b, v8f c) {
    return __builtin_amdgcn_wmma_f32_16x16x32_bf16(
        false, a.v, false, b.v, (short)0, c, false, false);
}

// LDS byte offset of a __shared__ object: flat LDS addresses keep the LDS
// offset in addr[31:0] (ISA 10.2 aperture mapping), so truncation is exact.
__device__ inline unsigned lds_offset(const void* p) {
    return (unsigned)(size_t)p;
}

// Async global->LDS copy of 16 bytes, GVS mode (SGPR base + 32-bit VGPR
// offset). Tracked by ASYNCcnt; completion via s_wait_asynccnt + barrier.
__device__ inline void async_b128(unsigned lds_byte, const void* gbase,
                                  unsigned gl_byte) {
    asm volatile("global_load_async_to_lds_b128 %0, %1, %2 offset:0"
                 :: "v"(lds_byte), "v"(gl_byte), "s"(gbase)
                 : "memory");
}
__device__ inline void wait_async_0() {
    asm volatile("s_wait_asynccnt 0x0" ::: "memory");
}
__device__ inline void wait_async_4() {
    asm volatile("s_wait_asynccnt 0x4" ::: "memory");
}

// ---------------------------------------------------------------------------
// Kernel 0: f32 [K,N] -> bf16 [N,K] tiled transpose (weight pre-conversion).
// ---------------------------------------------------------------------------
__global__ __launch_bounds__(256) void transpose_bf16_kernel(
    const float* __restrict__ W, unsigned short* __restrict__ Wt,
    int K, int N)
{
    __shared__ unsigned short t[64][65];
    const int k0 = blockIdx.y * 64;
    const int n0 = blockIdx.x * 64;
    for (int i = threadIdx.x; i < 4096; i += 256) {
        int r = i >> 6, c = i & 63;                 // r: k, c: n  (coalesced)
        t[c][r] = f32_to_bf16(W[(size_t)(k0 + r) * N + n0 + c]);
    }
    __syncthreads();
    for (int i = threadIdx.x; i < 4096; i += 256) {
        int r = i >> 6, c = i & 63;                 // r: n, c: k  (coalesced)
        Wt[(size_t)(n0 + r) * K + k0 + c] = t[r][c];
    }
}

// ---------------------------------------------------------------------------
// Kernel 1: mod = silu(condition) @ Wada + bada        [B, 6D]
// ---------------------------------------------------------------------------
__global__ __launch_bounds__(128) void mod_kernel(
    const float* __restrict__ cond, const float* __restrict__ Wada,
    const float* __restrict__ bada, float* __restrict__ mod)
{
    __shared__ float sil[DD];
    const int b = blockIdx.y;
    const int n = blockIdx.x * 128 + threadIdx.x;
    for (int k = threadIdx.x; k < DD; k += 128) {
        float c = cond[b * DD + k];
        sil[k] = c / (1.0f + __expf(-c));
    }
    __syncthreads();
    float acc = bada[n];
    for (int k = 0; k < DD; ++k)
        acc += sil[k] * Wada[(size_t)k * MODN + n];
    mod[(size_t)b * MODN + n] = acc;
}

// ---------------------------------------------------------------------------
// Kernel 2: per-row LayerNorm + adaLN modulate, emit bf16.
// ---------------------------------------------------------------------------
__global__ __launch_bounds__(256) void ln_mod_kernel(
    const float* __restrict__ x, const float* __restrict__ g,
    const float* __restrict__ be, const float* __restrict__ mod,
    int shift_col0, int scale_col0, unsigned short* __restrict__ out)
{
    __shared__ float r1[256], r2[256];
    const int row = blockIdx.x;
    const int b   = row / TT;
    const int tid = threadIdx.x;
    const float* xr = x + (size_t)row * DD;

    float s = 0.f, s2 = 0.f;
    for (int c = tid; c < DD; c += 256) {
        float v = xr[c];
        s += v; s2 += v * v;
    }
    r1[tid] = s; r2[tid] = s2;
    __syncthreads();
    for (int o = 128; o > 0; o >>= 1) {
        if (tid < o) { r1[tid] += r1[tid + o]; r2[tid] += r2[tid + o]; }
        __syncthreads();
    }
    const float mu  = r1[0] * (1.0f / DD);
    const float var = r2[0] * (1.0f / DD) - mu * mu;
    const float rs  = rsqrtf(var + EPS);

    const float* mrow = mod + (size_t)b * MODN;
    for (int c = tid; c < DD; c += 256) {
        float v  = (xr[c] - mu) * rs * g[c] + be[c];
        float sc = mrow[scale_col0 + c];
        float sh = mrow[shift_col0 + c];
        out[(size_t)row * DD + c] = f32_to_bf16(v * (1.0f + sc) + sh);
    }
}

// ---------------------------------------------------------------------------
// Kernel 3: bf16 WMMA GEMM, async double-buffered LDS staging.
//   A [M,K] bf16 row-major; Bt [N,K] bf16 row-major (pre-transposed weight).
//   Block tile 128x128x32, 8 waves, each wave 32x64 (2x4 WMMA C tiles).
//   mode 0: out(bf16)=acc   mode 1: out(f32)=resid+gate*(acc+bias)
//   mode 2: out(bf16)=gelu_exact(acc+bias)
// ---------------------------------------------------------------------------
#define GBM 128
#define GBN 128
#define GBK 32
#define STAGE_SH ((GBM + GBN) * GBK)     // shorts per stage (A then B)

__global__ __launch_bounds__(256) void gemm_bf16_kernel(
    const unsigned short* __restrict__ A,   // [M,K]
    const unsigned short* __restrict__ Bt,  // [N,K]
    int M, int N, int K,
    const float* __restrict__ bias,
    const float* __restrict__ mod, int gate_col0,
    const float* __restrict__ resid,
    void* __restrict__ out, int mode)
{
    __shared__ __align__(16) unsigned short smem[2 * STAGE_SH];   // 32 KB

    const int tid  = threadIdx.x;
    const int lane = tid & 31;
    const int wave = tid >> 5;
    const int hlf  = (lane >> 4) & 1;
    const int ln16 = lane & 15;

    const int bm0 = blockIdx.y * GBM;
    const int bn0 = blockIdx.x * GBN;
    const int wm0 = (wave >> 1) * 32;
    const int wn0 = (wave & 1) * 64;

    const unsigned lds0 = lds_offset(smem);
    const int nIter = K / GBK;

    // issue one stage of async copies: A tile 128x32, B tile 128x32 (8KB each)
    auto issue_stage = [&](int k0, int stage) {
        const unsigned sbase = lds0 + (unsigned)stage * STAGE_SH * 2;
#pragma unroll
        for (int p = 0; p < 2; ++p) {                 // A: 512 chunks / 256 thr
            int c = tid + p * 256;
            int r = c >> 2, kc = c & 3;
            async_b128(sbase + (unsigned)(r * GBK + kc * 8) * 2, A,
                       (unsigned)(((bm0 + r) * K) + k0 + kc * 8) * 2);
        }
        const unsigned bbase = sbase + (unsigned)(GBM * GBK) * 2;
#pragma unroll
        for (int p = 0; p < 2; ++p) {                 // B: 512 chunks / 256 thr
            int c = tid + p * 256;
            int r = c >> 2, kc = c & 3;
            async_b128(bbase + (unsigned)(r * GBK + kc * 8) * 2, Bt,
                       (unsigned)(((bn0 + r) * K) + k0 + kc * 8) * 2);
        }
    };

    v8f acc[2][4];
#pragma unroll
    for (int i = 0; i < 2; ++i)
#pragma unroll
        for (int j = 0; j < 4; ++j) acc[i][j] = v8f_zero();

    issue_stage(0, 0);

    for (int it = 0; it < nIter; ++it) {
        const int cur = it & 1;
        const bool pre = (it + 1) < nIter;
        if (pre) {
            issue_stage((it + 1) * GBK, cur ^ 1);
            wait_async_4();                 // keep prefetched stage in flight
        } else {
            wait_async_0();
        }
        __syncthreads();

        const unsigned short* As = smem + cur * STAGE_SH;
        const unsigned short* Bs = As + GBM * GBK;

        FragAB af[2], bf[4];
#pragma unroll
        for (int i = 0; i < 2; ++i) {
            const u32x4* rp = (const u32x4*)(As + (wm0 + i * 16 + ln16) * GBK);
            af[i].q[0] = rp[hlf];
            af[i].q[1] = rp[2 + hlf];
        }
#pragma unroll
        for (int j = 0; j < 4; ++j) {
            const u32x4* rp = (const u32x4*)(Bs + (wn0 + j * 16 + ln16) * GBK);
            bf[j].q[0] = rp[hlf];
            bf[j].q[1] = rp[2 + hlf];
        }
#pragma unroll
        for (int i = 0; i < 2; ++i)
#pragma unroll
            for (int j = 0; j < 4; ++j)
                acc[i][j] = wmma_bf16(af[i], bf[j], acc[i][j]);
        __syncthreads();
    }

    // epilogue; C layout: lane col n=ln16, reg r -> row r + 8*hlf
#pragma unroll
    for (int i = 0; i < 2; ++i) {
#pragma unroll
        for (int j = 0; j < 4; ++j) {
            const int ncol = bn0 + wn0 + j * 16 + ln16;
#pragma unroll
            for (int r = 0; r < 8; ++r) {
                const int mrow = bm0 + wm0 + i * 16 + r + 8 * hlf;
                float v = acc[i][j][r];
                if (bias) v += bias[ncol];
                if (mode == 1) {
                    const int bb = mrow / TT;
                    const float gt = mod[(size_t)bb * MODN + gate_col0 + ncol];
                    ((float*)out)[(size_t)mrow * N + ncol] =
                        resid[(size_t)mrow * N + ncol] + gt * v;
                } else if (mode == 2) {
                    const float ge = 0.5f * v * (1.0f + erff(v * 0.70710678118f));
                    ((unsigned short*)out)[(size_t)mrow * N + ncol] =
                        f32_to_bf16(ge);
                } else {
                    ((unsigned short*)out)[(size_t)mrow * N + ncol] =
                        f32_to_bf16(v);
                }
            }
        }
    }
}

// ---------------------------------------------------------------------------
// Kernel 4: causal flash attention, bf16 WMMA, async Q/K staging.
//   Grid (B*H, T/64); 128 threads = 4 waves; each wave owns 16 query rows.
// ---------------------------------------------------------------------------
__global__ __launch_bounds__(128) void attn_kernel(
    const unsigned short* __restrict__ qkv,  // [BT, 3*D] bf16
    unsigned short* __restrict__ aout)       // [BT, D]   bf16
{
    __shared__ __align__(16) unsigned short Qs[64][64];
    __shared__ __align__(16) unsigned short Ks[32][64];
    __shared__ __align__(16) unsigned short Vt[64][32];
    __shared__ __align__(16) unsigned short Ps[4][16][32];

    const int bh = blockIdx.x;
    const int b  = bh >> 4;
    const int h  = bh & 15;
    const int q0 = blockIdx.y * 64;

    const int tid  = threadIdx.x;
    const int lane = tid & 31;
    const int wave = tid >> 5;
    const int hlf  = (lane >> 4) & 1;
    const int ln16 = lane & 15;

    const unsigned ldsQ = lds_offset(Qs);
    const unsigned ldsK = lds_offset(Ks);

    // stage Q block 64x64 via async copies (4 chunks per thread)
    {
        const unsigned gq = (unsigned)(((b * TT + q0) * 3 * DD) + h * HDIM) * 2;
#pragma unroll
        for (int p = 0; p < 4; ++p) {
            int i = tid + p * 128;           // 512 chunks of 8 bf16
            int r = i >> 3, c = (i & 7) * 8;
            async_b128(ldsQ + (unsigned)(r * 64 + c) * 2, qkv,
                       gq + (unsigned)(r * 3 * DD + c) * 2);
        }
    }

    v8f ofr[4];
    float m_i[8], l_i[8];
#pragma unroll
    for (int d = 0; d < 4; ++d) ofr[d] = v8f_zero();
#pragma unroll
    for (int r = 0; r < 8; ++r) { m_i[r] = -1e30f; l_i[r] = 0.f; }

    const int kend = q0 + 64;
    for (int kb = 0; kb < kend; kb += 32) {
        __syncthreads();
        // K tile 32x64, async (2 chunks per thread)
        {
            const unsigned gk =
                (unsigned)(((b * TT + kb) * 3 * DD) + DD + h * HDIM) * 2;
#pragma unroll
            for (int p = 0; p < 2; ++p) {
                int i = tid + p * 128;       // 256 chunks
                int r = i >> 3, c = (i & 7) * 8;
                async_b128(ldsK + (unsigned)(r * 64 + c) * 2, qkv,
                           gk + (unsigned)(r * 3 * DD + c) * 2);
            }
        }
        // V tile transposed [d][k] (element-wise; async can't transpose)
        for (int i = tid; i < 2048; i += 128) {
            int r = i >> 6, c = i & 63;
            Vt[c][r] = qkv[(size_t)(b * TT + kb + r) * (3 * DD) + 2 * DD +
                           h * HDIM + c];
        }
        wait_async_0();
        __syncthreads();

        // ---- S = Q * K^T, two 16-key tiles ----
        v8f s[2];
#pragma unroll
        for (int nt = 0; nt < 2; ++nt) {
            v8f z = v8f_zero();
#pragma unroll
            for (int kc = 0; kc < 2; ++kc) {
                FragAB a, bb;
                const u32x4* ap = (const u32x4*)&Qs[wave * 16 + ln16][0];
                a.q[0] = ap[kc * 4 + hlf];
                a.q[1] = ap[kc * 4 + 2 + hlf];
                const u32x4* bp = (const u32x4*)&Ks[nt * 16 + ln16][0];
                bb.q[0] = bp[kc * 4 + hlf];
                bb.q[1] = bp[kc * 4 + 2 + hlf];
                z = wmma_bf16(a, bb, z);
            }
            s[nt] = z;
        }

        // ---- online softmax (query row = (reg r, hlf)) ----
        const float scale = 0.125f;
        float rmax[8];
#pragma unroll
        for (int r = 0; r < 8; ++r) rmax[r] = -1e30f;
#pragma unroll
        for (int nt = 0; nt < 2; ++nt) {
            const int key = kb + nt * 16 + ln16;
#pragma unroll
            for (int r = 0; r < 8; ++r) {
                const int q = q0 + wave * 16 + r + 8 * hlf;
                float v = s[nt][r] * scale;
                if (key > q) v = -1e30f;
                s[nt][r] = v;
                rmax[r] = fmaxf(rmax[r], v);
            }
        }
#pragma unroll
        for (int msk = 1; msk < 16; msk <<= 1)
#pragma unroll
            for (int r = 0; r < 8; ++r)
                rmax[r] = fmaxf(rmax[r], __shfl_xor(rmax[r], msk, 32));

        float sc[8], rsum[8];
#pragma unroll
        for (int r = 0; r < 8; ++r) {
            float mn = fmaxf(m_i[r], rmax[r]);
            sc[r] = __expf(m_i[r] - mn);
            m_i[r] = mn;
            rsum[r] = 0.f;
        }
#pragma unroll
        for (int nt = 0; nt < 2; ++nt)
#pragma unroll
            for (int r = 0; r < 8; ++r) {
                float p = __expf(s[nt][r] - m_i[r]);
                rsum[r] += p;
                Ps[wave][r + 8 * hlf][nt * 16 + ln16] = f32_to_bf16(p);
            }
#pragma unroll
        for (int msk = 1; msk < 16; msk <<= 1)
#pragma unroll
            for (int r = 0; r < 8; ++r)
                rsum[r] += __shfl_xor(rsum[r], msk, 32);
#pragma unroll
        for (int r = 0; r < 8; ++r) l_i[r] = l_i[r] * sc[r] + rsum[r];

#pragma unroll
        for (int d = 0; d < 4; ++d)
#pragma unroll
            for (int r = 0; r < 8; ++r) ofr[d][r] *= sc[r];

        FragAB pa;
        const u32x4* pp = (const u32x4*)&Ps[wave][ln16][0];
        pa.q[0] = pp[hlf];
        pa.q[1] = pp[2 + hlf];
#pragma unroll
        for (int d = 0; d < 4; ++d) {
            FragAB vb;
            const u32x4* vp = (const u32x4*)&Vt[d * 16 + ln16][0];
            vb.q[0] = vp[hlf];
            vb.q[1] = vp[2 + hlf];
            ofr[d] = wmma_bf16(pa, vb, ofr[d]);
        }
    }

#pragma unroll
    for (int r = 0; r < 8; ++r) l_i[r] = 1.0f / l_i[r];
#pragma unroll
    for (int d = 0; d < 4; ++d)
#pragma unroll
        for (int r = 0; r < 8; ++r) {
            const int row = b * TT + q0 + wave * 16 + r + 8 * hlf;
            const int col = h * HDIM + d * 16 + ln16;
            aout[(size_t)row * DD + col] = f32_to_bf16(ofr[d][r] * l_i[r]);
        }
}

// ---------------------------------------------------------------------------
// Host-side launcher
// ---------------------------------------------------------------------------
extern "C" void kernel_launch(void* const* d_in, const int* in_sizes, int n_in,
                              void* d_out, int out_size, void* d_ws, size_t ws_size,
                              hipStream_t stream) {
    (void)in_sizes; (void)n_in; (void)out_size; (void)ws_size;

    const float* x     = (const float*)d_in[0];
    const float* cond  = (const float*)d_in[1];
    const float* Wqkv  = (const float*)d_in[2];
    const float* Wproj = (const float*)d_in[3];
    const float* bproj = (const float*)d_in[4];
    const float* W1    = (const float*)d_in[5];
    const float* b1    = (const float*)d_in[6];
    const float* W2    = (const float*)d_in[7];
    const float* b2    = (const float*)d_in[8];
    const float* g1    = (const float*)d_in[9];
    const float* be1   = (const float*)d_in[10];
    const float* g2    = (const float*)d_in[11];
    const float* be2   = (const float*)d_in[12];
    const float* Wada  = (const float*)d_in[13];
    const float* bada  = (const float*)d_in[14];
    float* out = (float*)d_out;

    char* w = (char*)d_ws;
    auto take = [&](size_t bytes) {
        char* p = w;
        w += (bytes + 255) & ~(size_t)255;
        return p;
    };
    float*          mod   = (float*)         take((size_t)BB * MODN * 4);
    unsigned short* xm    = (unsigned short*)take((size_t)BT * DD * 2);
    unsigned short* qkv   = (unsigned short*)take((size_t)BT * 3 * DD * 2);
    unsigned short* attn  = (unsigned short*)take((size_t)BT * DD * 2);
    float*          x1    = (float*)         take((size_t)BT * DD * 4);
    unsigned short* hbuf  = (unsigned short*)take((size_t)BT * MLPH * 2);
    unsigned short* WqkvT = (unsigned short*)take((size_t)3 * DD * DD * 2);
    unsigned short* WprojT= (unsigned short*)take((size_t)DD * DD * 2);
    unsigned short* W1T   = (unsigned short*)take((size_t)DD * MLPH * 2);
    unsigned short* W2T   = (unsigned short*)take((size_t)MLPH * DD * 2);
    unsigned short* xm2   = xm;              // xm dead after QKV GEMM

    // 0. weight pre-conversion: f32 [K,N] -> bf16 [N,K]
    transpose_bf16_kernel<<<dim3(3 * DD / 64, DD / 64), 256, 0, stream>>>(
        Wqkv, WqkvT, DD, 3 * DD);
    transpose_bf16_kernel<<<dim3(DD / 64, DD / 64), 256, 0, stream>>>(
        Wproj, WprojT, DD, DD);
    transpose_bf16_kernel<<<dim3(MLPH / 64, DD / 64), 256, 0, stream>>>(
        W1, W1T, DD, MLPH);
    transpose_bf16_kernel<<<dim3(DD / 64, MLPH / 64), 256, 0, stream>>>(
        W2, W2T, MLPH, DD);

    // 1. modulation vector
    mod_kernel<<<dim3(MODN / 128, BB), 128, 0, stream>>>(cond, Wada, bada, mod);

    // 2. LN1 + modulate (shift=sh_a col 0, scale=sc_a col D)
    ln_mod_kernel<<<BT, 256, 0, stream>>>(x, g1, be1, mod, 0, DD, xm);

    // 3. QKV GEMM
    gemm_bf16_kernel<<<dim3(3 * DD / GBN, BT / GBM), 256, 0, stream>>>(
        xm, WqkvT, BT, 3 * DD, DD, nullptr, nullptr, 0, nullptr, qkv, 0);

    // 4. causal flash attention
    attn_kernel<<<dim3(BB * HH, TT / 64), 128, 0, stream>>>(qkv, attn);

    // 5. proj + gated residual
    gemm_bf16_kernel<<<dim3(DD / GBN, BT / GBM), 256, 0, stream>>>(
        attn, WprojT, BT, DD, DD, bproj, mod, 2 * DD, x, x1, 1);

    // 6. LN2 + modulate (shift=sh_m col 3D, scale=sc_m col 4D)
    ln_mod_kernel<<<BT, 256, 0, stream>>>(x1, g2, be2, mod, 3 * DD, 4 * DD, xm2);

    // 7. MLP up + exact GELU
    gemm_bf16_kernel<<<dim3(MLPH / GBN, BT / GBM), 256, 0, stream>>>(
        xm2, W1T, BT, MLPH, DD, b1, nullptr, 0, nullptr, hbuf, 2);

    // 8. MLP down + gated residual -> d_out
    gemm_bf16_kernel<<<dim3(DD / GBN, BT / GBM), 256, 0, stream>>>(
        hbuf, W2T, BT, DD, MLPH, b2, mod, 5 * DD, x1, out, 1);
}
